// LambdaRankLoss_29703993819991
// MI455X (gfx1250) — 8-bit matrix operand order repaired and checked by reading, compiled, not run on hardware
//
#include <hip/hip_runtime.h>
#include <hip/hip_bf16.h>
#include <math.h>

#define BQ 4
#define NN 160
#define TT 512

#define FP8_ONE 0x38

#define OFF_TAG   0
#define OFF_BM    (NN*TT)
#define OFF_P     (2*NN*TT)
#define OFF_MB    (3*NN*TT)
#define OFF_CU    (3*NN*TT + TT)
#define OFF_IML   (3*NN*TT + TT + NN*4)
#define PER_Q     (3*NN*TT + TT + 2*NN*4)

typedef __attribute__((ext_vector_type(16))) int   v16i;
typedef __attribute__((ext_vector_type(8)))  float v8f;

__global__ void zero_out_kernel(float* out) {
    if (threadIdx.x == 0) out[0] = 0.0f;
}

__global__ __launch_bounds__(256) void precompute_kernel(
    const float* __restrict__ Mf,
    const float* __restrict__ api,
    const int*   __restrict__ preds,
    unsigned char* __restrict__ ws)
{
    const int b   = blockIdx.x;
    const int tid = threadIdx.x;
    unsigned char* tag = ws + (size_t)b * PER_Q + OFF_TAG;
    unsigned char* bM  = ws + (size_t)b * PER_Q + OFF_BM;
    unsigned char* P   = ws + (size_t)b * PER_Q + OFF_P;
    unsigned char* Mb  = ws + (size_t)b * PER_Q + OFF_MB;
    float* cUiml = (float*)(ws + (size_t)b * PER_Q + OFF_CU);
    float* invML = (float*)(ws + (size_t)b * PER_Q + OFF_IML);

    __shared__ int   scnt[256];
    __shared__ float sMc;

    for (int t = tid; t < TT; t += 256)
        Mb[t] = (Mf[b * TT + t] != 0.0f) ? 1 : 0;
    __syncthreads();

    for (int idx = tid; idx < NN * TT; idx += 256) {
        int j = idx >> 9;
        int t = idx & (TT - 1);
        int pred = preds[b * NN + j];
        unsigned char tg = (api[(size_t)pred * TT + t] != 0.0f) ? 1 : 0;
        tag[idx] = tg;
        bM[idx]  = (unsigned char)((Mb[t] & (tg ^ 1)) ? FP8_ONE : 0);
    }
    __syncthreads();

    for (int t = tid; t < TT; t += 256) {
        unsigned char acc = 0;
        for (int j = 0; j < NN; ++j) {
            acc |= tag[j * TT + t];
            P[j * TT + t] = acc;
        }
    }

    int c = 0;
    for (int t = tid; t < TT; t += 256) c += Mb[t];
    scnt[tid] = c;
    __syncthreads();
    if (tid == 0) {
        int s = 0;
        for (int u = 0; u < 256; ++u) s += scnt[u];
        sMc = (float)(s + 1);
    }
    __syncthreads();

    for (int k = tid; k < NN; k += 256) {
        int cc = 0;
        for (int t = 0; t < TT; ++t) cc += Mb[t] & (P[k * TT + t] ^ 1);
        float im = 1.0f / (sMc * log2f((float)(k + 2)));
        invML[k] = im;
        cUiml[k] = (float)cc * im;
    }
}

__global__ __launch_bounds__(256) void lambda_main_kernel(
    const float* __restrict__ ys,
    const unsigned char* __restrict__ ws,
    float* __restrict__ out)
{
    const int bid  = blockIdx.x;
    const int b    = bid / (NN - 1);
    const int i    = bid % (NN - 1);
    const int tid  = threadIdx.x;
    const int lane = tid & 31;
    const int wave = tid >> 5;
    const int m    = lane & 15;
    const int h    = (lane >> 4) & 1;

    const unsigned char* tag = ws + (size_t)b * PER_Q + OFF_TAG;
    const unsigned char* bMg = ws + (size_t)b * PER_Q + OFF_BM;
    const unsigned char* P   = ws + (size_t)b * PER_Q + OFF_P;
    const unsigned char* Mb  = ws + (size_t)b * PER_Q + OFF_MB;
    const float* cUiml = (const float*)(ws + (size_t)b * PER_Q + OFF_CU);
    const float* invML = (const float*)(ws + (size_t)b * PER_Q + OFF_IML);

    __shared__ unsigned char aT[16 * 516];
    __shared__ float red[256];

    const int t0 = tid, t1 = tid + 256;
    const unsigned char mb0 = Mb[t0], mb1 = Mb[t1];
    unsigned char a0, a1;
    if (i == 1) { a0 = 0; a1 = 0; }
    else {
        int pr = (i == 0) ? (NN - 2) : (i - 2);
        a0 = P[pr * TT + t0];
        a1 = P[pr * TT + t1];
    }

    const float ysi = ys[b * NN + i];
    float lsum = 0.0f;

    const int nk  = NN - 1 - i;
    const int nRT = (nk + 15) >> 4;

    for (int rt = 0; rt < nRT; ++rt) {
        const int k0 = i + rt * 16;
        __syncthreads();
        for (int r = 0; r < 16; ++r) {
            int k = k0 + r;
            unsigned char v0 = 0, v1 = 0;
            if (k <= NN - 2) {
                v0 = (unsigned char)((mb0 & (a0 ^ 1)) ? FP8_ONE : 0);
                v1 = (unsigned char)((mb1 & (a1 ^ 1)) ? FP8_ONE : 0);
            }
            aT[r * 516 + t0] = v0;
            aT[r * 516 + t1] = v1;
            if (k <= NN - 2 && k > i) {
                a0 |= tag[k * TT + t0];
                a1 |= tag[k * TT + t1];
            }
        }
        __syncthreads();

        float cim[8], iml[8];
        #pragma unroll
        for (int r = 0; r < 8; ++r) {
            int   k  = k0 + r + 8 * h;
            int   kc = (k <= NN - 2) ? k : (NN - 2);
            float sc = (k <= NN - 2) ? 1.0f : 0.0f;
            cim[r] = cUiml[kc] * sc;
            iml[r] = invML[kc] * sc;
        }

        for (int ct = wave; ct < NN / 16; ct += 8) {
            const int j0 = ct * 16;
            if (j0 + 15 <= k0) continue;

            v8f dacc = {};
            #pragma unroll
            for (int c = 0; c < 4; ++c) {
                v16i A, Bv;
                #pragma unroll
                for (int v = 0; v < 16; ++v) {
                    int toff = c * 128 + (v >> 2) * 32 + h * 16 + (v & 3) * 4;
                    A[v] = *(const int*)(aT + m * 516 + toff);
                }
                #pragma unroll
                for (int v = 0; v < 16; ++v) {
                    int toff = c * 128 + (v >> 2) * 32 + h * 16 + (v & 3) * 4;
                    Bv[v] = *(const int*)(bMg + (size_t)(j0 + m) * TT + toff);
                }
                dacc = __builtin_amdgcn_wmma_f32_16x16x128_fp8_fp8(
                           A, Bv, (short)0, dacc, false, false);
            }

            const int j = j0 + m;
            const float lam =
                __builtin_amdgcn_rcpf(1.0f + __expf(ysi - ys[b * NN + j]));
            #pragma unroll
            for (int r = 0; r < 8; ++r) {
                int   k    = k0 + r + 8 * h;
                float term = cim[r] - dacc[r] * iml[r];
                term       = (j > k) ? term : 0.0f;
                lsum       = fmaf(lam, term, lsum);
            }
        }
    }

    red[tid] = lsum;
    __syncthreads();
    for (int s = 128; s > 0; s >>= 1) {
        if (tid < s) red[tid] += red[tid + s];
        __syncthreads();
    }
    if (tid == 0) {
        atomicAdd(out, red[0] * (1.0f / ((float)(NN + 1) * (float)BQ)));
    }
}

extern "C" void kernel_launch(void* const* d_in, const int* in_sizes, int n_in,
                              void* d_out, int out_size, void* d_ws, size_t ws_size,
                              hipStream_t stream) {
    (void)in_sizes; (void)n_in; (void)out_size; (void)ws_size;
    const float* ys   = (const float*)d_in[0];
    const float* Mf   = (const float*)d_in[1];
    const float* api  = (const float*)d_in[2];
    const int*   pred = (const int*)d_in[3];
    float* out = (float*)d_out;
    unsigned char* ws = (unsigned char*)d_ws;

    zero_out_kernel<<<1, 32, 0, stream>>>(out);
    precompute_kernel<<<BQ, 256, 0, stream>>>(Mf, api, pred, ws);
    lambda_main_kernel<<<BQ * (NN - 1), 256, 0, stream>>>(ys, ws, out);
}
